// get_model_54159537602741
// MI455X (gfx1250) — matrix operand order repaired and checked by reading, not compiled, not run on hardware
//
#include <hip/hip_runtime.h>
#include <hip/hip_bf16.h>
#include <math.h>
#include <stdint.h>

typedef _Float16 v16h __attribute__((ext_vector_type(16)));
typedef _Float16 h8   __attribute__((ext_vector_type(8)));
typedef float    v8f  __attribute__((ext_vector_type(8)));

#define LDS_Q(T) __attribute__((address_space(3))) T

#if defined(__has_builtin)
#  if __has_builtin(__builtin_amdgcn_tensor_load_to_lds) && __has_builtin(__builtin_amdgcn_s_wait_tensorcnt)
#    define USE_TDM 1
#  else
#    define USE_TDM 0
#  endif
#else
#  define USE_TDM 0
#endif

#define EPSBN 1e-5f
#define GEMM_T 128

// ---------------------------------------------------------------------------
// BN-fold: Wf[o,k] = half(W[o,k] * gamma[o]/sqrt(var+eps)) (zero-padded to kp)
// ---------------------------------------------------------------------------
__global__ void fold_bn(const float* __restrict__ W, const float* __restrict__ b,
                        const float* __restrict__ gamma, const float* __restrict__ beta,
                        const float* __restrict__ mean, const float* __restrict__ var,
                        _Float16* __restrict__ Wf, float* __restrict__ bf,
                        int cout, int cin, int kp)
{
    int t = blockIdx.x * blockDim.x + threadIdx.x;
    int total = cout * kp;
    if (t >= total) return;
    int o = t / kp;
    int k = t - o * kp;
    float scale = gamma[o] * rsqrtf(var[o] + EPSBN);
    Wf[t] = (k < cin) ? (_Float16)(W[(size_t)o * cin + k] * scale) : (_Float16)0.f;
    if (k == 0) bf[o] = (b[o] - mean[o]) * scale + beta[o];
}

// ---------------------------------------------------------------------------
// Farthest point sampling (one 256-thread block per batch; strided access)
// ---------------------------------------------------------------------------
#define FPS_T 256
__global__ void fps_kernel(const float* __restrict__ pts, long sb, long sn, long sc,
                           int Npts, int npoint,
                           int* __restrict__ fps_idx, float* __restrict__ new_xyz)
{
    __shared__ float dist[2048];
    __shared__ float rv[FPS_T];
    __shared__ int   ri[FPS_T];
    __shared__ int   far_s;

    int b = blockIdx.x;
    int t = threadIdx.x;
    const float* base = pts + (long)b * sb;

    for (int i = t; i < Npts; i += FPS_T) dist[i] = 1e10f;
    if (t == 0) far_s = 0;
    __syncthreads();

    for (int it = 0; it < npoint; ++it) {
        int far = far_s;
        float cx = base[(long)far * sn + 0 * sc];
        float cy = base[(long)far * sn + 1 * sc];
        float cz = base[(long)far * sn + 2 * sc];
        if (t == 0) {
            fps_idx[b * npoint + it] = far;
            float* o = new_xyz + ((long)b * npoint + it) * 3;
            o[0] = cx; o[1] = cy; o[2] = cz;
        }
        float bm = -1.f; int bi = 0;
        for (int i = t; i < Npts; i += FPS_T) {
            float dx = base[(long)i * sn + 0 * sc] - cx;
            float dy = base[(long)i * sn + 1 * sc] - cy;
            float dz = base[(long)i * sn + 2 * sc] - cz;
            float d  = dx * dx + dy * dy + dz * dz;
            float nd = fminf(dist[i], d);
            dist[i] = nd;
            if (nd > bm) { bm = nd; bi = i; }
        }
        rv[t] = bm; ri[t] = bi;
        __syncthreads();
        for (int off = FPS_T / 2; off > 0; off >>= 1) {
            if (t < off) {
                if (rv[t + off] > rv[t] ||
                    (rv[t + off] == rv[t] && ri[t + off] < ri[t])) {
                    rv[t] = rv[t + off]; ri[t] = ri[t + off];
                }
            }
            __syncthreads();
        }
        if (t == 0) far_s = ri[0];
        __syncthreads();
    }
}

// ---------------------------------------------------------------------------
// Ball query: one wave32 per query; ballot + prefix-popcount emits the first
// nsample in-radius indices in ascending order, padded with the first hit.
// ---------------------------------------------------------------------------
__global__ void __launch_bounds__(32)
ball_query_kernel(const float* __restrict__ pts, long sb, long sn, long sc, int Npts,
                  const float* __restrict__ centers, int S, int nsample, float r2,
                  int* __restrict__ out)
{
    __shared__ int sidx[64];
    int q = blockIdx.x;
    int b = q / S;
    int lane = threadIdx.x;
    const float* base = pts + (long)b * sb;
    float cx = centers[(long)q * 3 + 0];
    float cy = centers[(long)q * 3 + 1];
    float cz = centers[(long)q * 3 + 2];

    int cnt = 0;
    for (int b0 = 0; b0 < Npts && cnt < nsample; b0 += 32) {
        int i = b0 + lane;
        bool in = false;
        if (i < Npts) {
            float dx = base[(long)i * sn + 0 * sc] - cx;
            float dy = base[(long)i * sn + 1 * sc] - cy;
            float dz = base[(long)i * sn + 2 * sc] - cz;
            in = (dx * dx + dy * dy + dz * dz) <= r2;
        }
#if __has_builtin(__builtin_amdgcn_ballot_w32)
        unsigned mask = __builtin_amdgcn_ballot_w32(in);
#else
        unsigned mask = (unsigned)__ballot(in);
#endif
        int pos = cnt + __popc(mask & ((1u << lane) - 1u));
        if (in && pos < nsample) sidx[pos] = i;
        cnt += __popc(mask);
    }
    __syncthreads();
    if (cnt == 0) { if (lane == 0) sidx[0] = 0; cnt = 1; }
    __syncthreads();
    int valid = (cnt < nsample) ? cnt : nsample;
    int first = sidx[0];
    for (int j = lane; j < nsample; j += 32)
        out[(long)q * nsample + j] = (j < valid) ? sidx[j] : first;
}

// ---------------------------------------------------------------------------
// Grouping kernels: build f16 activation rows (K zero-padded to mult. of 32)
// ---------------------------------------------------------------------------
__global__ void group_sa1(const float* __restrict__ xyz, const float* __restrict__ new_xyz,
                          const int* __restrict__ ballidx, _Float16* __restrict__ act,
                          int N, int S, int ns, long rows)
{
    long r = (long)blockIdx.x * blockDim.x + threadIdx.x;
    if (r >= rows) return;
    long q = r / ns;
    int b = (int)(q / S);
    int idx = ballidx[r];
    const float* xb = xyz + (long)b * 6 * N;
    const float* c = new_xyz + q * 3;
    _Float16* o = act + r * 32;
    o[0] = (_Float16)(xb[0 * N + idx] - c[0]);
    o[1] = (_Float16)(xb[1 * N + idx] - c[1]);
    o[2] = (_Float16)(xb[2 * N + idx] - c[2]);
    o[3] = (_Float16)xb[3 * N + idx];
    o[4] = (_Float16)xb[4 * N + idx];
    o[5] = (_Float16)xb[5 * N + idx];
    for (int k = 6; k < 32; ++k) o[k] = (_Float16)0.f;
}

__global__ void group_sa2(const float* __restrict__ l1_xyz, const float* __restrict__ new_xyz,
                          const int* __restrict__ ballidx, const _Float16* __restrict__ l1_p,
                          _Float16* __restrict__ act,
                          int S1, int S, int ns, int C, int Kp, long rows)
{
    long r = (long)blockIdx.x * blockDim.x + threadIdx.x;
    if (r >= rows) return;
    long q = r / ns;
    int b = (int)(q / S);
    int idx = ballidx[r];
    long src = (long)b * S1 + idx;
    const float* ps = l1_xyz + src * 3;
    const float* c = new_xyz + q * 3;
    _Float16* o = act + r * Kp;
    o[0] = (_Float16)(ps[0] - c[0]);
    o[1] = (_Float16)(ps[1] - c[1]);
    o[2] = (_Float16)(ps[2] - c[2]);
    const _Float16* f = l1_p + src * C;
    for (int k = 0; k < C; ++k) o[3 + k] = f[k];
    for (int k = 3 + C; k < Kp; ++k) o[k] = (_Float16)0.f;
}

__global__ void group_sa3(const float* __restrict__ l2_xyz, const _Float16* __restrict__ l2_p,
                          _Float16* __restrict__ act, int C, int Kp, long rows)
{
    long r = (long)blockIdx.x * blockDim.x + threadIdx.x;
    if (r >= rows) return;
    _Float16* o = act + r * Kp;
    o[0] = (_Float16)l2_xyz[r * 3 + 0];
    o[1] = (_Float16)l2_xyz[r * 3 + 1];
    o[2] = (_Float16)l2_xyz[r * 3 + 2];
    const _Float16* f = l2_p + r * C;
    for (int k = 0; k < C; ++k) o[3 + k] = f[k];
    for (int k = 3 + C; k < Kp; ++k) o[k] = (_Float16)0.f;
}

// ---------------------------------------------------------------------------
// TDM helper: issue one 2D tile DMA (global -> LDS), 16-bit elements,
// tile = 32 (K) x 64 (rows), zero-fill past (dim0, dim1). D# per ISA §8.3/8.4.
// 6-arg builtin form on this toolchain.
// ---------------------------------------------------------------------------
#if USE_TDM
typedef unsigned int u32x4 __attribute__((ext_vector_type(4)));
typedef int i32x4 __attribute__((ext_vector_type(4)));
typedef int i32x8 __attribute__((ext_vector_type(8)));

__device__ __forceinline__ void tdm_load_tile(unsigned long long ga, unsigned lds_off,
                                              unsigned dim0, unsigned dim1,
                                              unsigned long long stride0)
{
    u32x4 g0;
    g0[0] = 1u;                                                  // count=1, user mode
    g0[1] = lds_off;                                             // lds_addr (bytes)
    g0[2] = (unsigned)ga;                                        // global_addr[31:0]
    g0[3] = ((unsigned)(ga >> 32) & 0x01FFFFFFu) | (2u << 30);   // addr[56:32] | type=2
    i32x8 g1;
    g1[0] = (int)(1u << 16);                                     // data_size = 2 bytes
    g1[1] = (int)((dim0 & 0xFFFFu) << 16);                       // tensor_dim0 lo
    g1[2] = (int)(((dim0 >> 16) & 0xFFFFu) | ((dim1 & 0xFFFFu) << 16));
    g1[3] = (int)(((dim1 >> 16) & 0xFFFFu) | (32u << 16));       // tile_dim0 = 32
    g1[4] = 64;                                                  // tile_dim1=64, tile_dim2=0
    g1[5] = (int)(unsigned)stride0;                              // tensor_dim0_stride lo
    g1[6] = (int)((unsigned)(stride0 >> 32) & 0xFFFFu);          // stride hi | dim1_stride lo
    g1[7] = 0;
    i32x4 z4 = {};                                               // groups 2/3: dims unused
    i32x8 z8 = {};
    __builtin_amdgcn_tensor_load_to_lds(g0, g1, z4, z4, z8, 0);
}
#endif

// ---------------------------------------------------------------------------
// WMMA GEMM: Out[r,c] = relu(sum_k A[r,k]*Wf[c,k] + bias[c])
// 128-thread (4-wave) workgroup computes a 64x64 tile. A/B slabs of K=32 are
// staged in LDS (double buffered) — by the Tensor Data Mover when available,
// else by cooperative vector loads. Wave w owns rows 16w..16w+15, all 64 cols.
// LDS is addressed through a laundered addrspace(3) offset pointer so the
// compiler (a) cannot prove it unwritten (TDM writes it behind the IR's back)
// and (b) still emits DS (not FLAT) accesses.
// ---------------------------------------------------------------------------
__global__ void __launch_bounds__(GEMM_T)
gemm_bias_relu(const _Float16* __restrict__ A, const _Float16* __restrict__ Wf,
               const float* __restrict__ bias, _Float16* __restrict__ Out,
               int K, int cout, long rows, int relu)
{
    struct SMem { _Float16 A[2][64 * 32]; _Float16 B[2][64 * 32]; };
    __shared__ SMem sm;
    // Anchor the LDS allocation (escapes so it is not elided).
    { SMem* anchor = &sm; asm volatile("" : "+s"(anchor)); }
    // Laundered LDS base offset (sole shared object -> offset 0).
    unsigned lds_base = 0;
    asm volatile("" : "+s"(lds_base));
    LDS_Q(_Float16)* L = (LDS_Q(_Float16)*)(unsigned long)lds_base;

    const int tid  = threadIdx.x;
    const int lane = tid & 31;
    const int wid  = tid >> 5;
    const int hsel = lane >> 4;
    const int l16  = lane & 15;
    const long rowBase = (long)blockIdx.x * 64;
    const int  colBase = blockIdx.y * 64;

    v8f acc0 = {}, acc1 = {}, acc2 = {}, acc3 = {};
    const int steps = K >> 5;

    // LDS half-element offsets: A[pb] @ pb*2048, B[pb] @ 4096 + pb*2048
#if USE_TDM
    const unsigned ldsA0 = 0u, ldsA1 = 4096u, ldsB0 = 8192u, ldsB1 = 12288u;  // bytes
    unsigned remM = (unsigned)(rows - rowBase);
    unsigned remN = (unsigned)(cout - colBase);
    if (tid < 32) {   // wave 0 drives the TDM (EXEC-independent, TENSORcnt-tracked)
        tdm_load_tile((unsigned long long)(uintptr_t)(A + rowBase * K),
                      ldsA0, (unsigned)K, remM, (unsigned long long)K);
        tdm_load_tile((unsigned long long)(uintptr_t)(Wf + (long)colBase * K),
                      ldsB0, (unsigned)K, remN, (unsigned long long)K);
    }
#else
    auto stage = [&](int pb, int k0) {
        for (int t = tid; t < 256; t += GEMM_T) {        // A: 64 rows x 32 halves
            int r = t >> 2, kk = (t & 3) * 8;
            h8 v = {};
            if (rowBase + r < rows) v = *(const h8*)(A + (rowBase + r) * (long)K + k0 + kk);
            *(LDS_Q(h8)*)(L + pb * 2048 + r * 32 + kk) = v;
        }
        for (int t = tid; t < 256; t += GEMM_T) {        // B: 64 cols x 32 halves
            int c = t >> 2, kk = (t & 3) * 8;
            *(LDS_Q(h8)*)(L + 4096 + pb * 2048 + c * 32 + kk) =
                *(const h8*)(Wf + (long)(colBase + c) * K + k0 + kk);
        }
    };
    stage(0, 0);
#endif

    for (int kb = 0; kb < steps; ++kb) {
        const int pb = kb & 1;
        const int k0n = (kb + 1) << 5;
#if USE_TDM
        if (tid < 32) {
            if (kb + 1 < steps) {   // prefetch next slab into the other buffer
                tdm_load_tile((unsigned long long)(uintptr_t)(A + rowBase * K + k0n),
                              (pb ? ldsA0 : ldsA1), (unsigned)(K - k0n), remM,
                              (unsigned long long)K);
                tdm_load_tile((unsigned long long)(uintptr_t)(Wf + (long)colBase * K + k0n),
                              (pb ? ldsB0 : ldsB1), (unsigned)(K - k0n), remN,
                              (unsigned long long)K);
                __builtin_amdgcn_s_wait_tensorcnt(2);   // current slab (2 oldest) done
            } else {
                __builtin_amdgcn_s_wait_tensorcnt(0);
            }
        }
        asm volatile("" ::: "memory");   // DMA-written LDS: force reloads below
#else
        if (kb + 1 < steps) stage(pb ^ 1, k0n);
#endif
        __syncthreads();

        // A fragment (16x32 f16, contiguous k = 16*hsel + i) from this wave's 16 staged rows
        const LDS_Q(_Float16)* As = L + pb * 2048 + (wid * 16 + l16) * 32 + 16 * hsel;
        h8 alo = *(const LDS_Q(h8)*)As;
        h8 ahi = *(const LDS_Q(h8)*)(As + 8);
        v16h a;
#pragma unroll
        for (int i = 0; i < 8; ++i) { a[i] = alo[i]; a[8 + i] = ahi[i]; }
        // B fragments: 16 contiguous K per half-wave, one column per lane
        const LDS_Q(_Float16)* Bb = L + 4096 + pb * 2048 + 16 * hsel;
        v16h vb0 = *(const LDS_Q(v16h)*)(Bb + (0 * 16 + l16) * 32);
        v16h vb1 = *(const LDS_Q(v16h)*)(Bb + (1 * 16 + l16) * 32);
        v16h vb2 = *(const LDS_Q(v16h)*)(Bb + (2 * 16 + l16) * 32);
        v16h vb3 = *(const LDS_Q(v16h)*)(Bb + (3 * 16 + l16) * 32);
        acc0 = __builtin_amdgcn_wmma_f32_16x16x32_f16(false, a, false, vb0, (short)0, acc0, false, false);
        acc1 = __builtin_amdgcn_wmma_f32_16x16x32_f16(false, a, false, vb1, (short)0, acc1, false, false);
        acc2 = __builtin_amdgcn_wmma_f32_16x16x32_f16(false, a, false, vb2, (short)0, acc2, false, false);
        acc3 = __builtin_amdgcn_wmma_f32_16x16x32_f16(false, a, false, vb3, (short)0, acc3, false, false);
        __syncthreads();
    }

    // Epilogue per C/D layout: VGPR v -> M = v + 8*hsel, N = l16
    float bi0 = bias[colBase + 0 * 16 + l16];
    float bi1 = bias[colBase + 1 * 16 + l16];
    float bi2 = bias[colBase + 2 * 16 + l16];
    float bi3 = bias[colBase + 3 * 16 + l16];
#pragma unroll
    for (int v = 0; v < 8; ++v) {
        long r = rowBase + wid * 16 + v + 8 * hsel;
        if (r < rows) {
            _Float16* orow = Out + r * (long)cout + colBase;
            float x0 = acc0[v] + bi0, x1 = acc1[v] + bi1;
            float x2 = acc2[v] + bi2, x3 = acc3[v] + bi3;
            if (relu) {
                x0 = fmaxf(x0, 0.f); x1 = fmaxf(x1, 0.f);
                x2 = fmaxf(x2, 0.f); x3 = fmaxf(x3, 0.f);
            }
            orow[0 * 16 + l16] = (_Float16)x0;
            orow[1 * 16 + l16] = (_Float16)x1;
            orow[2 * 16 + l16] = (_Float16)x2;
            orow[3 * 16 + l16] = (_Float16)x3;
        }
    }
}

// ---------------------------------------------------------------------------
// Max-pool over group dimension (optional extra fp32 copy for the l3 output)
// ---------------------------------------------------------------------------
__global__ void pool_max(const _Float16* __restrict__ act, _Float16* __restrict__ outh,
                         float* __restrict__ outf, int ns, int C, long outRows)
{
    long t = (long)blockIdx.x * blockDim.x + threadIdx.x;
    long total = outRows * C;
    if (t >= total) return;
    long r = t / C;
    int c = (int)(t - r * C);
    const _Float16* p = act + (r * ns) * (long)C + c;
    float m = -3.0e38f;
    for (int j = 0; j < ns; ++j) m = fmaxf(m, (float)p[(long)j * C]);
    outh[t] = (_Float16)m;
    if (outf) outf[t] = m;
}

// ---------------------------------------------------------------------------
// fc3 (256 -> 40) + log_softmax, one block (64 thr) per batch row
// ---------------------------------------------------------------------------
__global__ void __launch_bounds__(64)
fc3_logsoftmax(const _Float16* __restrict__ X, const float* __restrict__ W,
               const float* __restrict__ b, float* __restrict__ out)
{
    __shared__ float logits[40];
    __shared__ float sm[64];
    int row = blockIdx.x;
    int t = threadIdx.x;
    if (t < 40) {
        float s = b[t];
        const _Float16* xr = X + (long)row * 256;
        const float* wr = W + (long)t * 256;
        for (int k = 0; k < 256; ++k) s += (float)xr[k] * wr[k];
        logits[t] = s;
    }
    __syncthreads();
    sm[t] = (t < 40) ? logits[t] : -3.0e38f;
    __syncthreads();
    for (int off = 32; off > 0; off >>= 1) {
        if (t < off) sm[t] = fmaxf(sm[t], sm[t + off]);
        __syncthreads();
    }
    float mx = sm[0];
    __syncthreads();
    sm[t] = (t < 40) ? expf(logits[t] - mx) : 0.f;
    __syncthreads();
    for (int off = 32; off > 0; off >>= 1) {
        if (t < off) sm[t] += sm[t + off];
        __syncthreads();
    }
    float lse = logf(sm[0]);
    __syncthreads();
    if (t < 40) out[(long)row * 40 + t] = logits[t] - mx - lse;
}

// ---------------------------------------------------------------------------
// Host-side orchestration
// ---------------------------------------------------------------------------
extern "C" void kernel_launch(void* const* d_in, const int* in_sizes, int n_in,
                              void* d_out, int out_size, void* d_ws, size_t ws_size,
                              hipStream_t stream)
{
    (void)in_sizes; (void)n_in; (void)out_size; (void)ws_size;
    const int Bn = 16, Nn = 2048;
    const float* xyz = (const float*)d_in[0];

    // {cin, cout, Kp(padded), d_in base index}
    static const int LD[11][4] = {
        {   6,   64,   32,  1}, {  64,   64,   64,  7}, {  64,  128,   64, 13},
        { 131,  128,  160, 19}, { 128,  128,  128, 25}, { 128,  256,  128, 31},
        { 259,  256,  288, 37}, { 256,  512,  256, 43}, { 512, 1024,  512, 49},
        {1024,  512, 1024, 55}, { 512,  256,  512, 61},
    };

    char* wp = (char*)d_ws;
    auto carve = [&](size_t bytes) -> void* {
        void* p = (void*)wp;
        wp += (bytes + 255) & ~(size_t)255;
        return p;
    };

    _Float16* wf[11]; float* bfl[11];
    for (int i = 0; i < 11; ++i) {
        wf[i]  = (_Float16*)carve((size_t)LD[i][1] * LD[i][2] * sizeof(_Float16));
        bfl[i] = (float*)carve((size_t)LD[i][1] * sizeof(float));
    }
    int*   fps1  = (int*)carve((size_t)Bn * 512 * 4);
    float* nx1   = (float*)carve((size_t)Bn * 512 * 3 * 4);
    int*   ball1 = (int*)carve((size_t)Bn * 512 * 32 * 4);
    int*   fps2  = (int*)carve((size_t)Bn * 128 * 4);
    float* nx2   = (float*)carve((size_t)Bn * 128 * 3 * 4);
    int*   ball2 = (int*)carve((size_t)Bn * 128 * 64 * 4);
    _Float16* l1p  = (_Float16*)carve((size_t)Bn * 512 * 128 * 2);
    _Float16* l2p  = (_Float16*)carve((size_t)Bn * 128 * 256 * 2);
    _Float16* l3h  = (_Float16*)carve((size_t)Bn * 1024 * 2);
    _Float16* bufA = (_Float16*)carve((size_t)131072 * 160 * 2);
    _Float16* bufB = (_Float16*)carve((size_t)262144 * 128 * 2);

    for (int i = 0; i < 11; ++i) {
        int base = LD[i][3];
        int total = LD[i][1] * LD[i][2];
        fold_bn<<<(total + 255) / 256, 256, 0, stream>>>(
            (const float*)d_in[base + 0], (const float*)d_in[base + 1],
            (const float*)d_in[base + 2], (const float*)d_in[base + 3],
            (const float*)d_in[base + 4], (const float*)d_in[base + 5],
            wf[i], bfl[i], LD[i][1], LD[i][0], LD[i][2]);
    }

    auto gemm = [&](const _Float16* Ain, int li, _Float16* Op, long rows) {
        dim3 g((unsigned)((rows + 63) / 64), (unsigned)(LD[li][1] / 64));
        gemm_bias_relu<<<g, GEMM_T, 0, stream>>>(Ain, wf[li], bfl[li], Op,
                                                 LD[li][2], LD[li][1], rows, 1);
    };

    // ---- SA1 ----
    fps_kernel<<<Bn, FPS_T, 0, stream>>>(xyz, (long)6 * Nn, 1, Nn, Nn, 512, fps1, nx1);
    ball_query_kernel<<<Bn * 512, 32, 0, stream>>>(xyz, (long)6 * Nn, 1, Nn, Nn,
                                                   nx1, 512, 32, 0.04f, ball1);
    {
        long rows = (long)Bn * 512 * 32;   // 262144
        group_sa1<<<(unsigned)((rows + 255) / 256), 256, 0, stream>>>(
            xyz, nx1, ball1, bufA, Nn, 512, 32, rows);
        gemm(bufA, 0, bufB, rows);
        gemm(bufB, 1, bufA, rows);
        gemm(bufA, 2, bufB, rows);
        long orows = (long)Bn * 512;
        long total = orows * 128;
        pool_max<<<(unsigned)((total + 255) / 256), 256, 0, stream>>>(
            bufB, l1p, nullptr, 32, 128, orows);
    }

    // ---- SA2 ----
    fps_kernel<<<Bn, FPS_T, 0, stream>>>(nx1, (long)512 * 3, 3, 1, 512, 128, fps2, nx2);
    ball_query_kernel<<<Bn * 128, 32, 0, stream>>>(nx1, (long)512 * 3, 3, 1, 512,
                                                   nx2, 128, 64, 0.16f, ball2);
    {
        long rows = (long)Bn * 128 * 64;   // 131072
        group_sa2<<<(unsigned)((rows + 255) / 256), 256, 0, stream>>>(
            nx1, nx2, ball2, l1p, bufA, 512, 128, 64, 128, 160, rows);
        gemm(bufA, 3, bufB, rows);
        gemm(bufB, 4, bufA, rows);
        gemm(bufA, 5, bufB, rows);
        long orows = (long)Bn * 128;
        long total = orows * 256;
        pool_max<<<(unsigned)((total + 255) / 256), 256, 0, stream>>>(
            bufB, l2p, nullptr, 64, 256, orows);
    }

    // ---- SA3 (group all) ----
    {
        long rows = (long)Bn * 128;        // 2048
        group_sa3<<<(unsigned)((rows + 255) / 256), 256, 0, stream>>>(
            nx2, l2p, bufA, 256, 288, rows);
        gemm(bufA, 6, bufB, rows);
        gemm(bufB, 7, bufA, rows);
        gemm(bufA, 8, bufB, rows);
        long total = (long)Bn * 1024;
        pool_max<<<(unsigned)((total + 255) / 256), 256, 0, stream>>>(
            bufB, l3h, (float*)d_out + 640, 128, 1024, Bn);
    }

    // ---- FC head ----
    gemm(l3h, 9, bufA, Bn);    // 1024 -> 512
    gemm(bufA, 10, bufB, Bn);  // 512 -> 256
    fc3_logsoftmax<<<Bn, 64, 0, stream>>>(bufB, (const float*)d_in[67],
                                          (const float*)d_in[68], (float*)d_out);
}
